// SelfAttention_8976481649156
// MI455X (gfx1250) — matrix operand order repaired and checked by reading, not compiled, not run on hardware
//
#include <hip/hip_runtime.h>

// ---------- types ----------
typedef __bf16 bf16_t;
typedef __attribute__((ext_vector_type(2)))  __bf16 bf16x2;
typedef __attribute__((ext_vector_type(8)))  __bf16 v8bf;
typedef __attribute__((ext_vector_type(16))) __bf16 v16bf;
typedef __attribute__((ext_vector_type(8)))  float  v8f;
typedef __attribute__((ext_vector_type(4)))  unsigned int u32x4;
typedef __attribute__((ext_vector_type(8)))  int i32x8;
typedef __attribute__((ext_vector_type(4)))  int i32x4;

#define DIMV      1024
#define SEQ       2048
#define HEADS     16
#define DHEAD     64
#define INNER     1024
#define ROWS      4096          // b*n
#define QK_N      3072
#define LN_EPS    1e-5f

#if defined(__AMDGCN__) && __has_builtin(__builtin_amdgcn_tensor_load_to_lds)
#define HAVE_TDM 1
#else
#define HAVE_TDM 0
#endif

// ---------------------------------------------------------------------------
// TDM: 2D bf16 tile load global -> LDS (no transpose).  One wave issues it.
// D# packing per CDNA5 ISA ch.8 (group0 128b, group1 256b; groups 2/3 zero
// for 2D tensors).  data_size=1 (2 bytes).
// ---------------------------------------------------------------------------
__device__ __forceinline__ void tdm_load_tile_bf16(void* lds_dst, const void* gsrc,
                                                   int tile_rows, int tile_cols,
                                                   unsigned tensor_rows,
                                                   int stride_elems) {
#if HAVE_TDM
    unsigned long long ga = (unsigned long long)gsrc;
    unsigned lds_addr = (unsigned)(unsigned long long)lds_dst;  // LDS aperture: low 32b = offset
    u32x4 g0;
    g0[0] = 1u;                                                  // count=1, user descriptor
    g0[1] = lds_addr;                                            // lds_addr  (bits 63:32)
    g0[2] = (unsigned)ga;                                        // global_addr[31:0]
    g0[3] = (unsigned)((ga >> 32) & 0x01ffffffull) | (2u << 30); // global_addr[56:32] | type=2
    unsigned td0 = (unsigned)stride_elems;                       // tensor_dim0 (>= tile_cols)
    i32x8 g1;
    g1[0] = 1 << 16;                                             // data_size = 2 bytes
    g1[1] = (int)((td0 & 0xffffu) << 16);                        // tensor_dim0[15:0]  @ bits 63:48
    g1[2] = (int)((td0 >> 16) | ((tensor_rows & 0xffffu) << 16));// dim0 hi | tensor_dim1 lo
    g1[3] = (int)((tensor_rows >> 16) |
                  (((unsigned)tile_cols & 0xffffu) << 16));      // dim1 hi | tile_dim0
    g1[4] = tile_rows & 0xffff;                                  // tile_dim1 (tile_dim2 = 0)
    g1[5] = stride_elems;                                        // tensor_dim0_stride[31:0]
    g1[6] = 0;                                                   // stride hi | dim1_stride lo
    g1[7] = 0;
    i32x4 zz = {0, 0, 0, 0};
#if __clang_major__ >= 23
    i32x8 z8 = {0, 0, 0, 0, 0, 0, 0, 0};
    __builtin_amdgcn_tensor_load_to_lds(g0, g1, zz, zz, z8, 0);
#else
    __builtin_amdgcn_tensor_load_to_lds(g0, g1, zz, zz, 0);
#endif
#endif
}

__device__ __forceinline__ void tdm_wait() {
#if HAVE_TDM
    __builtin_amdgcn_s_wait_tensorcnt(0);
#endif
}

// ---------------------------------------------------------------------------
// WMMA helper:  D = A(16x32 bf16) * B(32x16 bf16) + C(16x16 f32)
// ---------------------------------------------------------------------------
__device__ __forceinline__ v8f wmma_bf16(v16bf a, v16bf b, v8f c) {
    return __builtin_amdgcn_wmma_f32_16x16x32_bf16(
        /*neg_a=*/false, a, /*neg_b=*/false, b,
        /*c_mod=*/(short)0, c, /*reuse_a=*/false, /*reuse_b=*/false);
}

// A-fragment gather (16x32, bf16) from row-major memory with row stride `ld`.
// gfx1250 layout: lanes 0-15 / 16-31 both hold rows M=0..15 (m = lane&15);
// VGPR pair v holds K = (v<4 ? 2v : 16+2(v-4)) + 8*(lane>>4) .. +1
__device__ __forceinline__ v16bf load_a_frag(const bf16_t* base, int ld, int lane) {
    int g = (lane >> 4) & 1;
    int m = lane & 15;
    const bf16_t* row = base + (size_t)m * ld;
    v16bf a;
#pragma unroll
    for (int v = 0; v < 8; ++v) {
        int kk = (v < 4) ? (2 * v + 8 * g) : (16 + 2 * (v - 4) + 8 * g);
        bf16x2 pr = *(const bf16x2*)(row + kk);
        a[2 * v]     = pr[0];
        a[2 * v + 1] = pr[1];
    }
    return a;
}

// B-fragment gather (32x16, bf16) from K-transposed storage:
// element (kk, n) lives at baseT[n*ld + kk].  Lane holds N = lane&15 and
// K = 8*(lane>>4) + e (e<8), 16 + 8*(lane>>4) + (e-8) (e>=8)  (2 x 16B loads).
__device__ __forceinline__ v16bf load_b_frag_T(const bf16_t* baseT, int ld, int lane) {
    int g = (lane >> 4) & 1;
    int n = lane & 15;
    const bf16_t* p = baseT + (size_t)n * ld + g * 8;
    v8bf lo = *(const v8bf*)(p);
    v8bf hi = *(const v8bf*)(p + 16);
    v16bf b;
#pragma unroll
    for (int i = 0; i < 8; ++i) { b[i] = lo[i]; b[8 + i] = hi[i]; }
    return b;
}

// ---------------------------------------------------------------------------
// Kernel 1: weight conversion fp32 -> bf16 (q scale 0.125 folded into w_qkv)
// ---------------------------------------------------------------------------
__global__ void conv_wqkv_kernel(const float* __restrict__ w, bf16_t* __restrict__ o) {
    size_t i = (size_t)blockIdx.x * 256 + threadIdx.x;
    if (i >= (size_t)DIMV * QK_N) return;
    int col = (int)(i % QK_N);
    float s = (col < INNER) ? 0.125f : 1.0f;   // DHEAD^-0.5 exactly
    o[i] = (bf16_t)(w[i] * s);
}

__global__ void conv_plain_kernel(const float* __restrict__ w, bf16_t* __restrict__ o, int n) {
    size_t i = (size_t)blockIdx.x * 256 + threadIdx.x;
    if (i < (size_t)n) o[i] = (bf16_t)w[i];
}

// ---------------------------------------------------------------------------
// Kernel 2: LayerNorm (fp32 in -> bf16 out), one block (256 thr) per row
// ---------------------------------------------------------------------------
__global__ void layernorm_kernel(const float* __restrict__ x,
                                 const float* __restrict__ gamma,
                                 const float* __restrict__ beta,
                                 bf16_t* __restrict__ xn) {
    __shared__ float red[16];
    int row = blockIdx.x;
    int t   = threadIdx.x;                 // 0..255, 8 waves
    const float* xr = x + (size_t)row * DIMV;
    float4 v = *(const float4*)(xr + t * 4);
    float s  = v.x + v.y + v.z + v.w;
    float s2 = v.x * v.x + v.y * v.y + v.z * v.z + v.w * v.w;
#pragma unroll
    for (int m = 1; m < 32; m <<= 1) {
        s  += __shfl_xor(s,  m, 32);
        s2 += __shfl_xor(s2, m, 32);
    }
    int wave = t >> 5, lane = t & 31;
    if (lane == 0) { red[wave] = s; red[8 + wave] = s2; }
    __syncthreads();
    float S = 0.f, S2 = 0.f;
#pragma unroll
    for (int w = 0; w < 8; ++w) { S += red[w]; S2 += red[8 + w]; }
    float mu  = S * (1.0f / DIMV);
    float var = S2 * (1.0f / DIMV) - mu * mu;
    float rs  = rsqrtf(var + LN_EPS);
    bf16_t* orow = xn + (size_t)row * DIMV + t * 4;
    const float* xv = &v.x;
#pragma unroll
    for (int e = 0; e < 4; ++e) {
        float g = gamma[t * 4 + e], b = beta[t * 4 + e];
        orow[e] = (bf16_t)((xv[e] - mu) * rs * g + b);
    }
}

// ---------------------------------------------------------------------------
// Kernels 3 & 5: tiled bf16 WMMA GEMM.  Block = 128 threads (4 waves),
// tile 64(M) x 64(N), BK = 32.  Each wave: 2x2 of 16x16 WMMA tiles.
// A tile staged via TDM (no transpose); B tile transposed via ds_store_b16.
// mode 0: split-store scaled QKV into [b,h,n,64] bf16; mode 1: fp32 + bias.
// ---------------------------------------------------------------------------
__global__ void gemm_bf16_kernel(const bf16_t* __restrict__ A,
                                 const bf16_t* __restrict__ Bw,
                                 int M, int N, int K, int mode,
                                 bf16_t* __restrict__ qo,
                                 bf16_t* __restrict__ ko,
                                 bf16_t* __restrict__ vo,
                                 float* __restrict__ outF,
                                 const float* __restrict__ bias) {
    __shared__ __align__(16) bf16_t As[64 * 32];   // [m][k]
    __shared__ __align__(16) bf16_t Bt[64 * 32];   // [n][k]  (transposed tile)

    int t    = threadIdx.x;            // 0..127
    int wave = t >> 5, lane = t & 31;
    int gm0  = blockIdx.y * 64;
    int gn0  = blockIdx.x * 64;
    int wm   = (wave >> 1) * 32;
    int wn   = (wave & 1)  * 32;

    v8f acc[2][2] = {{{}, {}}, {{}, {}}};

    for (int kb = 0; kb < K; kb += 32) {
#if HAVE_TDM
        if (wave == 0)
            tdm_load_tile_bf16(As, A + (size_t)gm0 * K + kb, 64, 32, (unsigned)M, K);
#else
        {   // A tile 64x32: 16 elems/thread, row = t/2, cols (t&1)*16 ..+15
            int ar = t >> 1, ac = (t & 1) * 16;
            const bf16_t* gA = A + (size_t)(gm0 + ar) * K + kb + ac;
            uint4 a0 = *(const uint4*)(gA);
            uint4 a1 = *(const uint4*)(gA + 8);
            *(uint4*)(As + ar * 32 + ac)     = a0;
            *(uint4*)(As + ar * 32 + ac + 8) = a1;
        }
#endif
        {   // B tile 32x64 -> transpose into Bt[n][k]; 16 elems/thread
            int br = t >> 2, bc = (t & 3) * 16;
            const bf16_t* gB = Bw + (size_t)(kb + br) * N + gn0 + bc;
            uint4 d0 = *(const uint4*)(gB);
            uint4 d1 = *(const uint4*)(gB + 8);
            const bf16_t* e0 = (const bf16_t*)&d0;
            const bf16_t* e1 = (const bf16_t*)&d1;
#pragma unroll
            for (int e = 0; e < 8; ++e) {
                Bt[(bc + e) * 32 + br]     = e0[e];
                Bt[(bc + 8 + e) * 32 + br] = e1[e];
            }
        }
        if (kb + 32 < K) {
            __builtin_prefetch(A + (size_t)(gm0 + (t >> 2)) * K + kb + 32, 0, 1);
            __builtin_prefetch(Bw + (size_t)(kb + 32 + (t & 31)) * N + gn0, 0, 1);
        }
#if HAVE_TDM
        if (wave == 0) tdm_wait();
#endif
        __syncthreads();

        v16bf a0 = load_a_frag(As + (wm +  0) * 32, 32, lane);
        v16bf a1 = load_a_frag(As + (wm + 16) * 32, 32, lane);
        v16bf b0 = load_b_frag_T(Bt + (wn +  0) * 32, 32, lane);
        v16bf b1 = load_b_frag_T(Bt + (wn + 16) * 32, 32, lane);
        acc[0][0] = wmma_bf16(a0, b0, acc[0][0]);
        acc[0][1] = wmma_bf16(a0, b1, acc[0][1]);
        acc[1][0] = wmma_bf16(a1, b0, acc[1][0]);
        acc[1][1] = wmma_bf16(a1, b1, acc[1][1]);
        __syncthreads();
    }

    int g = lane >> 4, n = lane & 15;
#pragma unroll
    for (int i = 0; i < 2; ++i)
#pragma unroll
        for (int j = 0; j < 2; ++j)
#pragma unroll
            for (int r = 0; r < 8; ++r) {
                int gm = gm0 + wm + i * 16 + r + 8 * g;
                int gn = gn0 + wn + j * 16 + n;
                float val = acc[i][j][r];
                if (mode == 0) {
                    int which  = gn >> 10;          // 0=q 1=k 2=v
                    int within = gn & 1023;
                    int h  = within >> 6, dh = within & 63;
                    int bb = gm >> 11,    nn = gm & 2047;
                    size_t off = ((((size_t)bb * HEADS + h) * SEQ) + nn) * DHEAD + dh;
                    bf16_t bv = (bf16_t)val;
                    if      (which == 0) qo[off] = bv;
                    else if (which == 1) ko[off] = bv;
                    else                 vo[off] = bv;
                } else {
                    outF[(size_t)gm * N + gn] = val + bias[gn];
                }
            }
}

// ---------------------------------------------------------------------------
// Kernel 4: flash attention.  grid = (SEQ/64, b*h), block = 128 (4 waves).
// Each wave owns 16 query rows; K tile staged via TDM, V tile transposed.
// ---------------------------------------------------------------------------
__global__ void attn_kernel(const bf16_t* __restrict__ Q,
                            const bf16_t* __restrict__ Kv,
                            const bf16_t* __restrict__ Vv,
                            const unsigned char* __restrict__ mask,
                            bf16_t* __restrict__ AO) {
    __shared__ __align__(16) bf16_t Ks[32 * 64];     // [key][d]
    __shared__ __align__(16) bf16_t Vt[64 * 32];     // [d][key]
    __shared__ __align__(16) bf16_t Ps[4][16 * 32];  // per-wave P staging

    int t    = threadIdx.x;
    int wave = t >> 5, lane = t & 31;
    int bh   = blockIdx.y;
    int b    = bh >> 4, h = bh & 15;
    size_t base = (size_t)bh * SEQ * DHEAD;
    int q0 = blockIdx.x * 64 + wave * 16;

    const bf16_t* qp = Q + base + (size_t)q0 * DHEAD;
    v16bf qa0 = load_a_frag(qp,      DHEAD, lane);   // d 0..31
    v16bf qa1 = load_a_frag(qp + 32, DHEAD, lane);   // d 32..63

    v8f o0 = {}, o1 = {}, o2 = {}, o3 = {};
    float mrow[8], lrow[8];
#pragma unroll
    for (int r = 0; r < 8; ++r) { mrow[r] = -3.0e38f; lrow[r] = 0.0f; }

    const unsigned char* mb = mask + (size_t)b * SEQ;
    int g = lane >> 4, nn = lane & 15;

    for (int j0 = 0; j0 < SEQ; j0 += 32) {
        __syncthreads();
#if HAVE_TDM
        if (wave == 0)
            tdm_load_tile_bf16(Ks, Kv + base + (size_t)j0 * DHEAD, 32, 64, SEQ, DHEAD);
#else
        {   // K tile 32x64 row-major: 16 elems/thread
            int kr = t >> 2, kc = (t & 3) * 16;
            const bf16_t* gK = Kv + base + (size_t)(j0 + kr) * DHEAD + kc;
            uint4 k0 = *(const uint4*)(gK);
            uint4 k1 = *(const uint4*)(gK + 8);
            *(uint4*)(Ks + kr * 64 + kc)     = k0;
            *(uint4*)(Ks + kr * 64 + kc + 8) = k1;
        }
#endif
        {   // V tile 32x64 -> transpose into Vt[d][key]
            int vr = t >> 2, vc = (t & 3) * 16;
            const bf16_t* gV = Vv + base + (size_t)(j0 + vr) * DHEAD + vc;
            uint4 d0 = *(const uint4*)(gV);
            uint4 d1 = *(const uint4*)(gV + 8);
            const bf16_t* e0 = (const bf16_t*)&d0;
            const bf16_t* e1 = (const bf16_t*)&d1;
#pragma unroll
            for (int e = 0; e < 8; ++e) {
                Vt[(vc + e) * 32 + vr]     = e0[e];
                Vt[(vc + 8 + e) * 32 + vr] = e1[e];
            }
        }
#if HAVE_TDM
        if (wave == 0) tdm_wait();
#endif
        __syncthreads();

        // S = Q * K^T : two 16x16 column tiles, K-dim 64 = 2 wmma steps each
        v8f s0 = {}, s1 = {};
        {
            v16bf kb00 = load_b_frag_T(Ks,                64, lane);
            v16bf kb01 = load_b_frag_T(Ks + 32,           64, lane);
            v16bf kb10 = load_b_frag_T(Ks + 16 * 64,      64, lane);
            v16bf kb11 = load_b_frag_T(Ks + 16 * 64 + 32, 64, lane);
            s0 = wmma_bf16(qa0, kb00, s0);
            s0 = wmma_bf16(qa1, kb01, s0);
            s1 = wmma_bf16(qa0, kb10, s1);
            s1 = wmma_bf16(qa1, kb11, s1);
        }

        // key mask (branchless, keeps EXEC all-ones around WMMA)
        float msk0 = (mb[j0 + nn]      != 0) ? 0.0f : -1.0e30f;
        float msk1 = (mb[j0 + 16 + nn] != 0) ? 0.0f : -1.0e30f;
#pragma unroll
        for (int r = 0; r < 8; ++r) { s0[r] += msk0; s1[r] += msk1; }

        // online softmax: each row spans one 16-lane group
#pragma unroll
        for (int r = 0; r < 8; ++r) {
            float mloc = fmaxf(s0[r], s1[r]);
#pragma unroll
            for (int xm = 1; xm < 16; xm <<= 1)
                mloc = fmaxf(mloc, __shfl_xor(mloc, xm, 32));
            float mnew  = fmaxf(mrow[r], mloc);
            float alpha = __expf(mrow[r] - mnew);
            float p0 = __expf(s0[r] - mnew);
            float p1 = __expf(s1[r] - mnew);
            float sum = p0 + p1;
#pragma unroll
            for (int xm = 1; xm < 16; xm <<= 1)
                sum += __shfl_xor(sum, xm, 32);
            lrow[r] = lrow[r] * alpha + sum;
            mrow[r] = mnew;
            s0[r] = p0; s1[r] = p1;
            o0[r] *= alpha; o1[r] *= alpha; o2[r] *= alpha; o3[r] *= alpha;
        }

        // C-layout P -> A-layout via per-wave LDS staging (same-wave DS: in order)
        bf16_t* Pw = Ps[wave];
#pragma unroll
        for (int r = 0; r < 8; ++r) {
            int M = r + 8 * g;
            Pw[M * 32 + nn]      = (bf16_t)s0[r];
            Pw[M * 32 + 16 + nn] = (bf16_t)s1[r];
        }
        v16bf pa = load_a_frag(Pw, 32, lane);

        // O += P * V  (4 output d-tiles)
        v16bf vb0 = load_b_frag_T(Vt +  0 * 32, 32, lane);
        v16bf vb1 = load_b_frag_T(Vt + 16 * 32, 32, lane);
        v16bf vb2 = load_b_frag_T(Vt + 32 * 32, 32, lane);
        v16bf vb3 = load_b_frag_T(Vt + 48 * 32, 32, lane);
        o0 = wmma_bf16(pa, vb0, o0);
        o1 = wmma_bf16(pa, vb1, o1);
        o2 = wmma_bf16(pa, vb2, o2);
        o3 = wmma_bf16(pa, vb3, o3);
    }

    // normalize and scatter into [b, n, h*64+dh] bf16 for the out-projection
#pragma unroll
    for (int r = 0; r < 8; ++r) {
        float inv = 1.0f / lrow[r];
        int row = b * SEQ + q0 + r + 8 * g;
        bf16_t* orow = AO + (size_t)row * INNER + h * DHEAD + nn;
        orow[ 0] = (bf16_t)(o0[r] * inv);
        orow[16] = (bf16_t)(o1[r] * inv);
        orow[32] = (bf16_t)(o2[r] * inv);
        orow[48] = (bf16_t)(o3[r] * inv);
    }
}

// ---------------------------------------------------------------------------
// launch
// ---------------------------------------------------------------------------
extern "C" void kernel_launch(void* const* d_in, const int* in_sizes, int n_in,
                              void* d_out, int out_size, void* d_ws, size_t ws_size,
                              hipStream_t stream) {
    const float*         x     = (const float*)d_in[0];
    const unsigned char* mask  = (const unsigned char*)d_in[1];
    const float*         gamma = (const float*)d_in[2];
    const float*         beta  = (const float*)d_in[3];
    const float*         wqkv  = (const float*)d_in[4];
    const float*         wout  = (const float*)d_in[5];
    const float*         bout  = (const float*)d_in[6];
    float*               out   = (float*)d_out;

    char* ws = (char*)d_ws;
    const size_t MB = 1024 * 1024;
    bf16_t* wqkv_bf = (bf16_t*)(ws +  0 * MB);   // 6 MB
    bf16_t* wout_bf = (bf16_t*)(ws +  6 * MB);   // 2 MB
    bf16_t* xn      = (bf16_t*)(ws +  8 * MB);   // 8 MB
    bf16_t* qb      = (bf16_t*)(ws + 16 * MB);   // 8 MB
    bf16_t* kb      = (bf16_t*)(ws + 24 * MB);   // 8 MB
    bf16_t* vb      = (bf16_t*)(ws + 32 * MB);   // 8 MB
    bf16_t* ao      = (bf16_t*)(ws + 40 * MB);   // 8 MB

    conv_wqkv_kernel<<<(DIMV * QK_N) / 256, 256, 0, stream>>>(wqkv, wqkv_bf);
    conv_plain_kernel<<<(DIMV * INNER) / 256, 256, 0, stream>>>(wout, wout_bf, DIMV * INNER);

    layernorm_kernel<<<ROWS, 256, 0, stream>>>(x, gamma, beta, xn);

    gemm_bf16_kernel<<<dim3(QK_N / 64, ROWS / 64), 128, 0, stream>>>(
        xn, wqkv_bf, ROWS, QK_N, DIMV, /*mode=*/0, qb, kb, vb, nullptr, nullptr);

    attn_kernel<<<dim3(SEQ / 64, 2 * HEADS), 128, 0, stream>>>(qb, kb, vb, mask, ao);

    gemm_bf16_kernel<<<dim3(INNER / 64, ROWS / 64), 128, 0, stream>>>(
        ao, wout_bf, ROWS, INNER, DIMV, /*mode=*/1, nullptr, nullptr, nullptr, out, bout);
}
